// SSMLP_49443663512208
// MI455X (gfx1250) — matrix operand order repaired and checked by reading, not compiled, not run on hardware
//
#include <hip/hip_runtime.h>
#include <stdint.h>

// Problem sizes (match reference)
#define NTOK 8192
#define BTOK 4096
#define HID  2048
#define FFN  8192

// LDS row stride (ushorts) for 32-K tiles: 32 data + 8 pad = 80 bytes
#define BSTRIDE 40

typedef __attribute__((ext_vector_type(16))) __bf16        v16bf;
typedef __attribute__((ext_vector_type(8)))  float         v8f;
typedef __attribute__((ext_vector_type(4)))  unsigned int  v4u;

struct B32Frag { v4u lo, hi; };   // 32 bytes == v16bf

#if __has_builtin(__builtin_amdgcn_sched_barrier)
#define SCHED_FENCE() __builtin_amdgcn_sched_barrier(0)
#else
#define SCHED_FENCE() asm volatile("" ::: "memory")
#endif

__device__ __forceinline__ unsigned short f2bf_raw(float f) {
  unsigned int u = __builtin_bit_cast(unsigned int, f);
  u += 0x7FFFu + ((u >> 16) & 1u);      // round-to-nearest-even
  return (unsigned short)(u >> 16);
}

// A fragment: 16x32 bf16, row-major source X[ld], rows row0..row0+15, K k0..k0+31.
// ISA layout: lane l -> row (l&15); K = { (l>>4)*8 + 0..7 , 16 + (l>>4)*8 + 0..7 }
__device__ __forceinline__ v16bf load_frag_a(const unsigned short* __restrict__ X,
                                             int ld, int row0, int k0) {
  int lane = threadIdx.x & 31;
  int m = lane & 15;
  int h = lane >> 4;
  const unsigned short* p = X + (size_t)(row0 + m) * ld + (k0 + h * 8);
  B32Frag t;
  t.lo = *(const v4u*)p;           // K = k0 + h*8 + 0..7
  t.hi = *(const v4u*)(p + 16);    // K = k0 + 16 + h*8 + 0..7
  return __builtin_bit_cast(v16bf, t);
}

// B fragment from an LDS tile stored as [64 rows(N)][BSTRIDE ushorts of K].
// lane l -> col (l&15) within 16-wide subtile t; K = { (l>>4)*8 + 0..7 , 16 + (l>>4)*8 + 0..7 }
__device__ __forceinline__ v16bf load_frag_b_lds(const unsigned short* base, int subtile) {
  int lane = threadIdx.x & 31;
  int n = lane & 15;
  int h = lane >> 4;
  const unsigned short* p = base + (subtile * 16 + n) * BSTRIDE + h * 8;
  B32Frag t;
  t.lo = *(const v4u*)p;
  t.hi = *(const v4u*)(p + 16);
  return __builtin_bit_cast(v16bf, t);
}

// CDNA5 async copy: 16 bytes global -> LDS per lane, tracked by ASYNCcnt.
__device__ __forceinline__ void async_copy_b128(unsigned short* lds_dst,
                                                const unsigned short* gsrc) {
  unsigned lds_addr = (unsigned)(uintptr_t)lds_dst;   // low 32 bits = LDS byte offset
  asm volatile("global_load_async_to_lds_b128 %0, %1, off"
               :: "v"(lds_addr), "v"(gsrc) : "memory");
}

__device__ __forceinline__ void wait_async_all() {
  asm volatile("s_wait_asynccnt 0x0" ::: "memory");
}

// ---- Kernel 1: gather routed tokens and cast fp32 -> bf16 -------------------
__global__ void k_gather_cast(const float* __restrict__ hs,
                              const long long* __restrict__ idx,
                              unsigned short* __restrict__ xbf) {
  int b = blockIdx.x;
  long long src = idx[b];
  const float4* in4 = (const float4*)(hs + (size_t)src * HID) + threadIdx.x * 2;
  float4 a = in4[0];
  float4 c = in4[1];
  uint4 o;
  o.x = (unsigned)f2bf_raw(a.x) | ((unsigned)f2bf_raw(a.y) << 16);
  o.y = (unsigned)f2bf_raw(a.z) | ((unsigned)f2bf_raw(a.w) << 16);
  o.z = (unsigned)f2bf_raw(c.x) | ((unsigned)f2bf_raw(c.y) << 16);
  o.w = (unsigned)f2bf_raw(c.z) | ((unsigned)f2bf_raw(c.w) << 16);
  ((uint4*)(xbf + (size_t)b * HID))[threadIdx.x] = o;
}

// ---- Kernel 2: cast fp32 [R,C] -> bf16 transposed [C,R] (LDS 32x33 tile) ----
__global__ void k_cast_transpose(const float* __restrict__ in,
                                 unsigned short* __restrict__ out,
                                 int R, int C) {
  __shared__ unsigned short tile[32][33];
  int c = blockIdx.x * 32 + threadIdx.x;
  int rbase = blockIdx.y * 32;
#pragma unroll
  for (int i = 0; i < 4; ++i) {
    int rl = threadIdx.y + i * 8;
    tile[rl][threadIdx.x] = f2bf_raw(in[(size_t)(rbase + rl) * C + c]);
  }
  __syncthreads();
  int r = rbase + threadIdx.x;
#pragma unroll
  for (int i = 0; i < 4; ++i) {
    int cl = threadIdx.y + i * 8;
    out[(size_t)(blockIdx.x * 32 + cl) * R + r] = tile[threadIdx.x][cl];
  }
}

// ---- Kernel 3: h = bf16( silu(x@W1) * (x@W3) ) ------------------------------
// Block: 256 threads = 8 waves; block tile 128 rows x 64 cols; wave tile 16x64.
// B tiles (W1 + W3, 64 cols x 32 K each) double-buffered in LDS via async copy.
__global__ void __launch_bounds__(256)
k_gemm1_swiglu(const unsigned short* __restrict__ xbf,
               const unsigned short* __restrict__ w1t,
               const unsigned short* __restrict__ w3t,
               unsigned short* __restrict__ hbf) {
  __shared__ unsigned short ldsB[2][2][64 * BSTRIDE];  // [buf][mat g/u][row*BSTRIDE+k]

  int wave = threadIdx.x >> 5;
  int row0 = blockIdx.y * 128 + wave * 16;
  int col0 = blockIdx.x * 64;

  // staging role: thread -> (row 0..63, 16B chunk 0..3) for both matrices
  int srow = threadIdx.x >> 2;
  int sch  = (threadIdx.x & 3) * 8;     // ushort offset of 16B chunk

  v8f accg[4] = {};
  v8f accu[4] = {};

  // prologue: stage k=0, preload A(k=0)
  async_copy_b128(&ldsB[0][0][srow * BSTRIDE + sch],
                  w1t + (size_t)(col0 + srow) * HID + sch);
  async_copy_b128(&ldsB[0][1][srow * BSTRIDE + sch],
                  w3t + (size_t)(col0 + srow) * HID + sch);
  v16bf a = load_frag_a(xbf, HID, row0, 0);
  wait_async_all();
  __syncthreads();

  const int nk = HID / 32;
  for (int kk = 0; kk < nk; ++kk) {
    int cur = kk & 1;

    // issue next-iteration staging + A load first (overlap with compute below)
    v16bf a_next = a;
    if (kk + 1 < nk) {
      int knext = (kk + 1) * 32;
      async_copy_b128(&ldsB[cur ^ 1][0][srow * BSTRIDE + sch],
                      w1t + (size_t)(col0 + srow) * HID + knext + sch);
      async_copy_b128(&ldsB[cur ^ 1][1][srow * BSTRIDE + sch],
                      w3t + (size_t)(col0 + srow) * HID + knext + sch);
      a_next = load_frag_a(xbf, HID, row0, knext);
    }
    SCHED_FENCE();

    // phase 1: issue ALL fragment loads (single ds clause, 16 loads in flight)
    v16bf bg[4], bu[4];
#pragma unroll
    for (int t = 0; t < 4; ++t) {
      bg[t] = load_frag_b_lds(&ldsB[cur][0][0], t);
      bu[t] = load_frag_b_lds(&ldsB[cur][1][0], t);
    }
    SCHED_FENCE();   // loads may not sink past here; WMMAs may not hoist above

    // phase 2: 8 back-to-back WMMAs (partial dscnt waits only)
#pragma unroll
    for (int t = 0; t < 4; ++t) {
      accg[t] = __builtin_amdgcn_wmma_f32_16x16x32_bf16(
          false, a, false, bg[t], (short)0, accg[t], false, false);
      accu[t] = __builtin_amdgcn_wmma_f32_16x16x32_bf16(
          false, a, false, bu[t], (short)0, accu[t], false, false);
    }

    a = a_next;
    wait_async_all();
    __syncthreads();
  }

  int lane = threadIdx.x & 31;
  int n = lane & 15;
  int h = lane >> 4;
#pragma unroll
  for (int t = 0; t < 4; ++t) {
#pragma unroll
    for (int r = 0; r < 8; ++r) {
      float g = accg[t][r];
      float u = accu[t][r];
      float s = g / (1.0f + __expf(-g));     // silu(g)
      int row = row0 + h * 8 + r;            // D layout: VGPR r -> M = r + 8*(lane>>4)
      int col = col0 + 16 * t + n;
      hbf[(size_t)row * FFN + col] = f2bf_raw(s * u);
    }
  }
}

// ---- Kernel 4: out = rw * (h @ W2) ------------------------------------------
__global__ void __launch_bounds__(256)
k_gemm2_down(const unsigned short* __restrict__ hbf,
             const unsigned short* __restrict__ w2t,
             const float* __restrict__ rw,
             float* __restrict__ out) {
  __shared__ unsigned short ldsB[2][64 * BSTRIDE];

  int wave = threadIdx.x >> 5;
  int row0 = blockIdx.y * 128 + wave * 16;
  int col0 = blockIdx.x * 64;

  int srow = threadIdx.x >> 2;
  int sch  = (threadIdx.x & 3) * 8;

  v8f acc[4] = {};

  async_copy_b128(&ldsB[0][srow * BSTRIDE + sch],
                  w2t + (size_t)(col0 + srow) * FFN + sch);
  v16bf a = load_frag_a(hbf, FFN, row0, 0);
  wait_async_all();
  __syncthreads();

  const int nk = FFN / 32;
  for (int kk = 0; kk < nk; ++kk) {
    int cur = kk & 1;

    v16bf a_next = a;
    if (kk + 1 < nk) {
      int knext = (kk + 1) * 32;
      async_copy_b128(&ldsB[cur ^ 1][srow * BSTRIDE + sch],
                      w2t + (size_t)(col0 + srow) * FFN + knext + sch);
      a_next = load_frag_a(hbf, FFN, row0, knext);
    }
    SCHED_FENCE();

    v16bf b[4];
#pragma unroll
    for (int t = 0; t < 4; ++t) {
      b[t] = load_frag_b_lds(&ldsB[cur][0], t);
    }
    SCHED_FENCE();

#pragma unroll
    for (int t = 0; t < 4; ++t) {
      acc[t] = __builtin_amdgcn_wmma_f32_16x16x32_bf16(
          false, a, false, b[t], (short)0, acc[t], false, false);
    }

    a = a_next;
    wait_async_all();
    __syncthreads();
  }

  int lane = threadIdx.x & 31;
  int n = lane & 15;
  int h = lane >> 4;
#pragma unroll
  for (int r = 0; r < 8; ++r) {
    int row = row0 + h * 8 + r;
    float s = rw[row];
#pragma unroll
    for (int t = 0; t < 4; ++t) {
      out[(size_t)row * HID + col0 + 16 * t + n] = s * acc[t][r];
    }
  }
}

extern "C" void kernel_launch(void* const* d_in, const int* in_sizes, int n_in,
                              void* d_out, int out_size, void* d_ws, size_t ws_size,
                              hipStream_t stream) {
  (void)in_sizes; (void)n_in; (void)out_size; (void)ws_size;

  const float*     hs  = (const float*)d_in[0];      // [NTOK, HID] fp32
  const long long* idx = (const long long*)d_in[1];  // [BTOK] int64
  const float*     rw  = (const float*)d_in[2];      // [BTOK, 1] fp32
  const float*     W1  = (const float*)d_in[3];      // [HID, FFN] fp32
  const float*     W3  = (const float*)d_in[4];      // [HID, FFN] fp32
  const float*     W2  = (const float*)d_in[5];      // [FFN, HID] fp32
  float*           out = (float*)d_out;              // [BTOK, HID] fp32

  // Workspace layout (bf16 elements), total ~184.5 MB
  char* ws = (char*)d_ws;
  unsigned short* xbf = (unsigned short*)ws;                      // [BTOK, HID]
  unsigned short* w1t = (unsigned short*)(ws + (size_t)BTOK * HID * 2);
  unsigned short* w3t = w1t + (size_t)FFN * HID;                  // [FFN, HID]
  unsigned short* w2t = w3t + (size_t)FFN * HID;                  // [HID, FFN]
  unsigned short* hbf = w2t + (size_t)HID * FFN;                  // [BTOK, FFN]

  // 1) gather + cast tokens
  k_gather_cast<<<BTOK, 256, 0, stream>>>(hs, idx, xbf);

  // 2) weight cast + transpose to [N, K] bf16
  dim3 tb(32, 8);
  k_cast_transpose<<<dim3(FFN / 32, HID / 32), tb, 0, stream>>>(W1, w1t, HID, FFN);
  k_cast_transpose<<<dim3(FFN / 32, HID / 32), tb, 0, stream>>>(W3, w3t, HID, FFN);
  k_cast_transpose<<<dim3(HID / 32, FFN / 32), tb, 0, stream>>>(W2, w2t, FFN, HID);

  // 3) fused gate/up GEMM + SwiGLU -> h (bf16)
  k_gemm1_swiglu<<<dim3(FFN / 64, BTOK / 128), 256, 0, stream>>>(xbf, w1t, w3t, hbf);

  // 4) down GEMM + routing-weight scale -> out (fp32)
  k_gemm2_down<<<dim3(HID / 64, BTOK / 128), 256, 0, stream>>>(hbf, w2t, rw, out);
}
